// WindowAttention_9500467659156
// MI455X (gfx1250) — matrix operand order repaired and checked by reading, not compiled, not run on hardware
//
#include <hip/hip_runtime.h>
#include <hip/hip_bf16.h>

typedef __attribute__((ext_vector_type(16))) _Float16 v16h;
typedef __attribute__((ext_vector_type(8)))  _Float16 v8h;
typedef __attribute__((ext_vector_type(8)))  float    v8f;

#define WMMA_F16(a, b, c) \
  __builtin_amdgcn_wmma_f32_16x16x32_f16(false, (a), false, (b), (short)0, (c), false, false)

// ---------------- fragment loaders (CDNA5 16-bit WMMA VGPR layouts) ----------------
// A (16xK tile, row-major source, ld in elements): lane 0-15 -> M=lane, halves 0..7 = K kb+0..7,
// halves 8..15 = K kb+16..23 ; lanes 16-31 shifted by +8 in K.
__device__ __forceinline__ v16h load_frag_a(const _Float16* src, int row0, int ld, int kb, int lane) {
  int m  = row0 + (lane & 15);
  int ko = kb + ((lane >> 4) << 3);
  const _Float16* p = src + (size_t)m * ld + ko;
  union { v16h v; v8h h[2]; } u;
  u.h[0] = *(const v8h*)(p);
  u.h[1] = *(const v8h*)(p + 16);
  return u.v;
}
// B (Kx16 tile). Source is column-major (column n has contiguous K, stride ld between columns):
// lane 0-15 -> N=lane, halves 0..7 = K kb+0..7, halves 8..15 = K kb+16..23 ; lanes 16-31 shifted by +8 in K.
__device__ __forceinline__ v16h load_frag_b(const _Float16* srcT, int col0, int ld, int kb, int lane) {
  int n = col0 + (lane & 15);
  int k = kb + ((lane >> 4) << 3);
  const _Float16* p = srcT + (size_t)n * ld + k;
  union { v16h v; v8h h[2]; } u;
  u.h[0] = *(const v8h*)(p);
  u.h[1] = *(const v8h*)(p + 16);
  return u.v;
}

// ---------------- constants / workspace layout ----------------
#define HEADS 12
#define HD    32
#define WS7   7
#define NWIN  64           // 8x8 windows
#define TOKP  64           // 49 tokens padded to 64
#define B_    16
#define DIM   384
#define NQKV  1152
#define MPAD  (B_ * NWIN * TOKP)      // 65536 padded rows
#define MROWS (B_ * 56 * 56)          // 50176 real rows

// byte offsets in d_ws (all multiples of 256)
#define OFF_WTQKV  ((size_t)0)                        // 1152*384 f16
#define OFF_WTOUT  ((size_t)884736)                   // 384*384 f16
#define OFF_WTPOST ((size_t)1179648)                  // 384*384 f16
#define OFF_AWIN   ((size_t)1474560)                  // 65536*384 f16 ; reused as merged-O (50176*384 f16)
#define OFF_Q      ((size_t)51806208)                 // 16*12*64*64*32 f16 ; reused as Y1 (50176*384 f16)
#define OFF_K      ((size_t)102137856)
#define OFF_V      ((size_t)152469504)

// ---------------- 32x64 strip GEMM core: double-buffered, 8 WMMAs per K-step ----------------
// acc[mt][nt] for mt in {0,1} (rows row0, row0+16), nt in 0..3 (cols n0 + 16*nt). KDIM = 384.
__device__ __forceinline__ void gemm_strip_2x4(const _Float16* __restrict__ A,
                                               const _Float16* __restrict__ Wt,
                                               int row0, int n0, int lda, int lane,
                                               v8f acc[2][4]) {
  v16h a0 = load_frag_a(A, row0,      lda, 0, lane);
  v16h a1 = load_frag_a(A, row0 + 16, lda, 0, lane);
  v16h b[4];
#pragma unroll
  for (int t = 0; t < 4; ++t) b[t] = load_frag_b(Wt, n0 + t * 16, DIM, 0, lane);
#pragma unroll
  for (int kb = 0; kb + 32 < DIM; kb += 32) {
    // prefetch next K-step while the 8 WMMAs below execute
    v16h a0n = load_frag_a(A, row0,      lda, kb + 32, lane);
    v16h a1n = load_frag_a(A, row0 + 16, lda, kb + 32, lane);
    v16h bn[4];
#pragma unroll
    for (int t = 0; t < 4; ++t) bn[t] = load_frag_b(Wt, n0 + t * 16, DIM, kb + 32, lane);
#pragma unroll
    for (int t = 0; t < 4; ++t) {
      acc[0][t] = WMMA_F16(a0, b[t], acc[0][t]);
      acc[1][t] = WMMA_F16(a1, b[t], acc[1][t]);
    }
    a0 = a0n; a1 = a1n;
#pragma unroll
    for (int t = 0; t < 4; ++t) b[t] = bn[t];
  }
#pragma unroll
  for (int t = 0; t < 4; ++t) {           // epilogue K-step
    acc[0][t] = WMMA_F16(a0, b[t], acc[0][t]);
    acc[1][t] = WMMA_F16(a1, b[t], acc[1][t]);
  }
}

// ---------------- prep kernels ----------------
__global__ void k_prep_w(const float* wqkv, const float* wout, const float* wpost,
                         _Float16* tq, _Float16* to, _Float16* tp) {
  int stride = gridDim.x * blockDim.x;
  for (int i = blockIdx.x * blockDim.x + threadIdx.x; i < DIM * NQKV; i += stride) {
    int k = i / NQKV, n = i % NQKV;
    tq[(size_t)n * DIM + k] = (_Float16)wqkv[i];
  }
  for (int i = blockIdx.x * blockDim.x + threadIdx.x; i < DIM * DIM; i += stride) {
    int k = i / DIM, n = i % DIM;
    to[(size_t)n * DIM + k] = (_Float16)wout[i];
    tp[(size_t)n * DIM + k] = (_Float16)wpost[i];
  }
}

// rolled (-3,-3) + window-partitioned + token-padded activation matrix, fp32 -> f16
__global__ void k_prep_awin(const float* x, _Float16* A) {
  long total = (long)MPAD * DIM;
  long stride = (long)gridDim.x * blockDim.x;
  for (long i = blockIdx.x * (long)blockDim.x + threadIdx.x; i < total; i += stride) {
    int c = (int)(i % DIM);
    long t = i / DIM;
    int tok = (int)(t & 63);
    _Float16 val = (_Float16)0.0f;
    if (tok < 49) {
      int win = (int)((t >> 6) & 63);
      int b   = (int)(t >> 12);
      int wy = win >> 3, wx = win & 7;
      int ty = tok / WS7, tx = tok % WS7;
      int y  = (wy * WS7 + ty + 3) % 56;
      int xx = (wx * WS7 + tx + 3) % 56;
      val = (_Float16)x[((size_t)(b * 56 + y) * 56 + xx) * DIM + c];
    }
    A[i] = val;
  }
}

// ---------------- QKV GEMM: (65536 x 384) @ (384 x 1152), scatter into q/k/v ----------------
__global__ void __launch_bounds__(32) k_gemm_qkv(const _Float16* __restrict__ A,
                                                 const _Float16* __restrict__ Wt,
                                                 _Float16* q, _Float16* k, _Float16* v) {
  int lane = threadIdx.x;
  int row0 = blockIdx.x * 32;
  int n0   = blockIdx.y * 64;
  v8f acc[2][4] = {};
  gemm_strip_2x4(A, Wt, row0, n0, DIM, lane, acc);

  const float qscale = 0.17677669529663687f;   // 32^-0.5 folded into q
#pragma unroll
  for (int mt = 0; mt < 2; ++mt) {
#pragma unroll
    for (int t = 0; t < 4; ++t) {
      int n = n0 + t * 16 + (lane & 15);
      int which = n / DIM, rem = n % DIM;
      int head = rem >> 5, d = rem & 31;
#pragma unroll
      for (int r = 0; r < 8; ++r) {
        int trow = row0 + mt * 16 + r + ((lane >> 4) << 3);
        int tok = trow & 63, win = (trow >> 6) & 63, b = trow >> 12;
        size_t base = ((size_t)(b * HEADS + head) * NWIN + win) * (TOKP * HD);
        float val = acc[mt][t][r];
        if (which == 0)       q[base + (size_t)tok * HD + d] = (_Float16)(val * qscale);
        else if (which == 1)  k[base + (size_t)tok * HD + d] = (_Float16)val;
        else                  v[base + (size_t)d * TOKP + tok] = (_Float16)val;  // transposed
      }
    }
  }
}

// ---------------- attention: one wave per (b, head, window) ----------------
__global__ void __launch_bounds__(32) k_attn(const _Float16* __restrict__ qg,
                                             const _Float16* __restrict__ kg,
                                             const _Float16* __restrict__ vg,
                                             const float* __restrict__ pe,
                                             _Float16* __restrict__ outm) {
  __shared__ __align__(16) float    S[64 * 65];
  __shared__ __align__(16) _Float16 P[64 * 64];
  __shared__ float pes[169];
  int lane = threadIdx.x;
  int job  = blockIdx.x;                           // (b*12 + head)*64 + win
  const _Float16* q = qg + (size_t)job * (TOKP * HD);
  const _Float16* k = kg + (size_t)job * (TOKP * HD);
  const _Float16* v = vg + (size_t)job * (TOKP * HD);
  for (int i = lane; i < 169; i += 32) pes[i] = pe[i];
  int win = job & 63, head = (job >> 6) % HEADS, b = job / (NWIN * HEADS);
  int wy = win >> 3, wx = win & 7;
  bool mrow = (wy == 7), mcol = (wx == 7);
  __syncthreads();

  // S = q @ k^T (+bias +mask) — 16 WMMAs, K = head_dim = 32 in one instruction each.
  // Preload all K-tiles of k^T as B fragments so WMMAs run back-to-back.
  v16h bk[4];
#pragma unroll
  for (int nt = 0; nt < 4; ++nt) bk[nt] = load_frag_b(k, nt * 16, HD, 0, lane);
  for (int mt = 0; mt < 4; ++mt) {
    v16h a = load_frag_a(q, mt * 16, HD, 0, lane);
    for (int nt = 0; nt < 4; ++nt) {
      v8f c = {};
      c = WMMA_F16(a, bk[nt], c);
      int j = nt * 16 + (lane & 15);
#pragma unroll
      for (int r = 0; r < 8; ++r) {
        int i = mt * 16 + r + ((lane >> 4) << 3);
        float val = -1e30f;
        if (i < 49 && j < 49) {
          int xi = i / WS7, yi = i % WS7, xj = j / WS7, yj = j % WS7;
          val = c[r] + pes[(xj - xi + 6) * 13 + (yj - yi + 6)];
          if (mrow && ((xi >= 4) != (xj >= 4))) val = -1e30f;
          if (mcol && ((yi >= 4) != (yj >= 4))) val = -1e30f;
        }
        S[i * 65 + j] = val;
      }
    }
  }
  __syncthreads();

  // softmax: 2 rows per lane
  for (int rr = 0; rr < 2; ++rr) {
    int i = lane + rr * 32;
    float mx = -1e30f;
    for (int j = 0; j < 64; ++j) mx = fmaxf(mx, S[i * 65 + j]);
    float sum = 0.f;
    for (int j = 0; j < 64; ++j) sum += __expf(S[i * 65 + j] - mx);
    float inv = 1.f / sum;
    for (int j = 0; j < 64; ++j) P[i * 64 + j] = (_Float16)(__expf(S[i * 65 + j] - mx) * inv);
  }
  __syncthreads();

  // O = P @ V  (K = 64 padded -> 2 chained WMMAs), store into merged (b,H,W,384) layout
  v16h bv[2][2];
#pragma unroll
  for (int nt = 0; nt < 2; ++nt)
#pragma unroll
    for (int kc = 0; kc < 2; ++kc) bv[nt][kc] = load_frag_b(v, nt * 16, TOKP, kc * 32, lane);
  for (int mt = 0; mt < 4; ++mt) {
    v16h a0 = load_frag_a((const _Float16*)P, mt * 16, 64, 0, lane);
    v16h a1 = load_frag_a((const _Float16*)P, mt * 16, 64, 32, lane);
    for (int nt = 0; nt < 2; ++nt) {
      v8f c = {};
      c = WMMA_F16(a0, bv[nt][0], c);
      c = WMMA_F16(a1, bv[nt][1], c);
      int col = head * HD + nt * 16 + (lane & 15);
#pragma unroll
      for (int r = 0; r < 8; ++r) {
        int i = mt * 16 + r + ((lane >> 4) << 3);
        if (i < 49) {
          int y  = wy * WS7 + i / WS7;
          int xx = wx * WS7 + i % WS7;
          outm[((size_t)(b * 3136 + y * 56 + xx)) * DIM + col] = (_Float16)c[r];
        }
      }
    }
  }
}

// ---------------- projection 1: Y1 = A @ Wout + b_out (f16 out) ----------------
__global__ void __launch_bounds__(32) k_gemm_out(const _Float16* __restrict__ A,
                                                 const _Float16* __restrict__ Wt,
                                                 const float* __restrict__ bias,
                                                 _Float16* __restrict__ Y) {
  int lane = threadIdx.x;
  int row0 = blockIdx.x * 32;
  int n0   = blockIdx.y * 64;
  v8f acc[2][4] = {};
  gemm_strip_2x4(A, Wt, row0, n0, DIM, lane, acc);
#pragma unroll
  for (int mt = 0; mt < 2; ++mt) {
#pragma unroll
    for (int t = 0; t < 4; ++t) {
      int n = n0 + t * 16 + (lane & 15);
      float bv = bias[n];
#pragma unroll
      for (int r = 0; r < 8; ++r) {
        int m = row0 + mt * 16 + r + ((lane >> 4) << 3);
        Y[(size_t)m * DIM + n] = (_Float16)(acc[mt][t][r] + bv);
      }
    }
  }
}

// ---------------- projection 2: out = roll(+3,+3)(Y1 @ Wpost + b_post) (fp32 out) ----------------
__global__ void __launch_bounds__(32) k_gemm_post(const _Float16* __restrict__ A,
                                                  const _Float16* __restrict__ Wt,
                                                  const float* __restrict__ bias,
                                                  float* __restrict__ out) {
  int lane = threadIdx.x;
  int row0 = blockIdx.x * 32;
  int n0   = blockIdx.y * 64;
  v8f acc[2][4] = {};
  gemm_strip_2x4(A, Wt, row0, n0, DIM, lane, acc);
#pragma unroll
  for (int mt = 0; mt < 2; ++mt) {
#pragma unroll
    for (int t = 0; t < 4; ++t) {
      int n = n0 + t * 16 + (lane & 15);
      float bv = bias[n];
#pragma unroll
      for (int r = 0; r < 8; ++r) {
        int m = row0 + mt * 16 + r + ((lane >> 4) << 3);
        int b = m / 3136, s = m % 3136;
        int y = (s / 56 + 3) % 56, xx = (s % 56 + 3) % 56;   // inverse roll
        out[((size_t)(b * 3136 + y * 56 + xx)) * DIM + n] = acc[mt][t][r] + bv;
      }
    }
  }
}

extern "C" void kernel_launch(void* const* d_in, const int* in_sizes, int n_in,
                              void* d_out, int out_size, void* d_ws, size_t ws_size,
                              hipStream_t stream) {
  const float* x      = (const float*)d_in[0];
  const float* w_qkv  = (const float*)d_in[1];
  const float* pe     = (const float*)d_in[2];
  const float* w_out  = (const float*)d_in[3];
  const float* b_out  = (const float*)d_in[4];
  const float* w_post = (const float*)d_in[5];
  const float* b_post = (const float*)d_in[6];
  float* out = (float*)d_out;

  char* ws = (char*)d_ws;
  _Float16* wt_qkv  = (_Float16*)(ws + OFF_WTQKV);
  _Float16* wt_out  = (_Float16*)(ws + OFF_WTOUT);
  _Float16* wt_post = (_Float16*)(ws + OFF_WTPOST);
  _Float16* a_win   = (_Float16*)(ws + OFF_AWIN);
  _Float16* q_scr   = (_Float16*)(ws + OFF_Q);
  _Float16* k_scr   = (_Float16*)(ws + OFF_K);
  _Float16* v_scr   = (_Float16*)(ws + OFF_V);
  _Float16* merged  = (_Float16*)(ws + OFF_AWIN);  // reuse: A_win dead after QKV GEMM
  _Float16* y1      = (_Float16*)(ws + OFF_Q);     // reuse: q dead after attention

  hipLaunchKernelGGL(k_prep_w, dim3(512), dim3(256), 0, stream,
                     w_qkv, w_out, w_post, wt_qkv, wt_out, wt_post);
  hipLaunchKernelGGL(k_prep_awin, dim3(8192), dim3(256), 0, stream, x, a_win);
  hipLaunchKernelGGL(k_gemm_qkv, dim3(MPAD / 32, NQKV / 64), dim3(32), 0, stream,
                     a_win, wt_qkv, q_scr, k_scr, v_scr);
  hipLaunchKernelGGL(k_attn, dim3(B_ * HEADS * NWIN), dim3(32), 0, stream,
                     q_scr, k_scr, v_scr, pe, merged);
  hipLaunchKernelGGL(k_gemm_out, dim3(MROWS / 32, DIM / 64), dim3(32), 0, stream,
                     merged, wt_out, b_out, y1);
  hipLaunchKernelGGL(k_gemm_post, dim3(MROWS / 32, DIM / 64), dim3(32), 0, stream,
                     y1, wt_post, b_post, out);
}
